// SDT_BiGRU_Classifier_78889959293205
// MI455X (gfx1250) — matrix operand order repaired and checked by reading, not compiled, not run on hardware
//
#include <hip/hip_runtime.h>
#include <hip/hip_bf16.h>

// Problem constants (from reference)
#define BATCH 2048
#define NSEQ  20
#define DMODEL 512
#define NHEAD 8
#define DKH   64
#define KSZ   3
#define DQF   64
#define ROWS  (BATCH * NSEQ)          // 40960
#define KCONV (KSZ * DMODEL)          // 1536
#define NT_TILES (DMODEL / 16)        // 32 column tiles
#define KK_CONV (KCONV / 32)          // 48 k-chunks for conv GEMM
#define KK_OUT  (DMODEL / 32)         // 16 k-chunks for out GEMM
#define MROWS 32                      // rows per workgroup (2 WMMA M-tiles)

typedef __attribute__((ext_vector_type(16))) _Float16 v16h;
typedef __attribute__((ext_vector_type(8)))  float    v8f;

// ---------------------------------------------------------------------------
// Pack conv weights W[o, i, tap] (f32) into WMMA-B fragment order (f16):
//   dst[((nt*48 + kk)*32 + lane)*16 + j]
//   B fragment (32x16, KxN, f16): n = lane&15 ; k = (j>>3)*16 + (lane>>4)*8 + (j&7)
//   K axis of the GEMM = tap*512 + i  (im2col order, matches A staging)
// ---------------------------------------------------------------------------
__global__ void pack_qkv_kernel(const float* __restrict__ Wq,
                                const float* __restrict__ Wk,
                                const float* __restrict__ Wv,
                                _Float16* __restrict__ pq,
                                _Float16* __restrict__ pk,
                                _Float16* __restrict__ pv) {
  int p = blockIdx.y;
  const float* W = (p == 0) ? Wq : (p == 1) ? Wk : Wv;
  _Float16* dst  = (p == 0) ? pq : (p == 1) ? pk : pv;
  size_t idx = (size_t)blockIdx.x * 256 + threadIdx.x;   // < 32*48*32*16
  int j    = (int)(idx & 15);
  int lane = (int)((idx >> 4) & 31);
  int kk   = (int)((idx >> 9) % 48);
  int nt   = (int)(idx / 24576);
  int o = nt * 16 + (lane & 15);
  int kelem = ((j >> 3) << 4) + ((lane >> 4) << 3) + (j & 7);          // 0..31
  int K = kk * 32 + kelem;                     // 0..1535
  int tap = K >> 9;
  int i = K & 511;
  dst[idx] = (_Float16)W[((size_t)o * DMODEL + i) * KSZ + tap];
}

// Pack Wo[i, o] into fragment order: dst[((nt*16 + kk)*32 + lane)*16 + j]
__global__ void pack_wo_kernel(const float* __restrict__ Wo,
                               _Float16* __restrict__ dst) {
  size_t idx = (size_t)blockIdx.x * 256 + threadIdx.x;   // < 32*16*32*16
  int j    = (int)(idx & 15);
  int lane = (int)((idx >> 4) & 31);
  int kk   = (int)((idx >> 9) & 15);
  int nt   = (int)(idx / 8192);
  int o = nt * 16 + (lane & 15);
  int K = kk * 32 + ((j >> 3) << 4) + ((lane >> 4) << 3) + (j & 7);    // 0..511 == input channel i
  dst[idx] = (_Float16)Wo[(size_t)K * DMODEL + o];
}

// ---------------------------------------------------------------------------
// qf = tanh(x @ Wqf + bqf);  s[b,n,h] = qf . Wqp[:,h]
// One wave per row; 8 rows per 256-thread block.
// ---------------------------------------------------------------------------
__global__ void __launch_bounds__(256)
qf_kernel(const float* __restrict__ x, const float* __restrict__ Wqf,
          const float* __restrict__ bqf, const float* __restrict__ Wqp,
          float* __restrict__ sbuf) {
  __shared__ float qfs[8][DQF];
  int wv = threadIdx.x >> 5, lane = threadIdx.x & 31;
  size_t gr = (size_t)blockIdx.x * 8 + wv;
  const float* xr = x + gr * DMODEL;
  float a0 = bqf[lane], a1 = bqf[lane + 32];
  for (int i = 0; i < DMODEL; ++i) {
    float xv = xr[i];
    a0 += xv * Wqf[(size_t)i * DQF + lane];
    a1 += xv * Wqf[(size_t)i * DQF + lane + 32];
  }
  qfs[wv][lane]      = tanhf(a0);
  qfs[wv][lane + 32] = tanhf(a1);
  __syncthreads();
  if (lane < NHEAD) {
    float s = 0.0f;
    for (int d = 0; d < DQF; ++d) s += qfs[wv][d] * Wqp[d * NHEAD + lane];
    sbuf[gr * NHEAD + lane] = s;
  }
}

// ---------------------------------------------------------------------------
// Fused conv-projection GEMM (WMMA f16) + LayerNorm + residual -> f16.
// 32-row tile per workgroup (2 WMMA M-tiles sharing each B fragment),
// 8 waves x 4 column tiles, 48 K-chunks of 32.
// Dynamic LDS: A tile (32x1536 f16, 96KB) reused as C tile (32x512 f32, 64KB).
// B-fragment addresses strength-reduced: one pointer bump per K-chunk,
// compile-time immediate offsets per column tile.
// ---------------------------------------------------------------------------
__global__ void __launch_bounds__(256)
proj_kernel(const float* __restrict__ x, const _Float16* __restrict__ wp,
            const float* __restrict__ gamma, const float* __restrict__ beta,
            _Float16* __restrict__ dst) {
  extern __shared__ __align__(16) char smem[];          // 96KB dynamic
  _Float16 (*As)[KCONV] = (_Float16(*)[KCONV])smem;
  float (*Cs)[DMODEL]   = (float(*)[DMODEL])smem;       // reused after GEMM
  __shared__ float redS[MROWS][8], redQ[MROWS][8], mu[MROWS], inv[MROWS];

  int tid = threadIdx.x;
  size_t tile = blockIdx.x;

  // ---- stage im2col A tile (zero-padded conv taps), f32 -> f16 ----
  for (int v8 = tid; v8 < MROWS * KCONV / 8; v8 += 256) {
    int r = v8 / (KCONV / 8);
    int c = (v8 - r * (KCONV / 8)) * 8;
    int tap = c >> 9;
    int i = c & 511;
    size_t gr = tile * MROWS + r;
    int bb = (int)(gr / NSEQ);
    int nn = (int)(gr - (size_t)bb * NSEQ);
    int n2 = nn + tap - 1;
    union { _Float16 h[8]; uint4 u; } tmp;
    if (n2 >= 0 && n2 < NSEQ) {
      const float* sp = x + ((size_t)(bb * NSEQ + n2) * DMODEL + i);
#pragma unroll
      for (int e = 0; e < 8; ++e) tmp.h[e] = (_Float16)sp[e];
    } else {
#pragma unroll
      for (int e = 0; e < 8; ++e) tmp.h[e] = (_Float16)0.0f;
    }
    *(uint4*)&As[r][c] = tmp.u;
  }
  __syncthreads();

  int w = tid >> 5, lane = tid & 31;         // 8 waves
  int m = lane & 15;
  int ksel = (lane >> 4) << 3;               // 0 or 8 (A-fragment K select)
  v8f acc[2][4] = {};

  // per-wave base pointers; advance by constant strides only
  const _Float16* ap0 = &As[m][ksel];
  const _Float16* ap1 = &As[16 + m][ksel];
  const _Float16* bptr = wp + (((size_t)(w * 4) * KK_CONV) * 32 + lane) * 16;

  for (int kk = 0; kk < KK_CONV; ++kk) {
    v16h a0, a1;
    *(uint4*)&a0       = *(const uint4*)ap0;          // K = ksel+0..7
    *((uint4*)&a0 + 1) = *(const uint4*)(ap0 + 16);   // K = 16+ksel+0..7
    *(uint4*)&a1       = *(const uint4*)ap1;
    *((uint4*)&a1 + 1) = *(const uint4*)(ap1 + 16);
#pragma unroll
    for (int t = 0; t < 4; ++t) {
      // byte offset t*49152 is a compile-time immediate (fits 24-bit IOFFSET)
      v16h bf = *(const v16h*)(bptr + (size_t)t * (KK_CONV * 512));
      acc[0][t] = __builtin_amdgcn_wmma_f32_16x16x32_f16(
          false, a0, false, bf, (short)0, acc[0][t], false, false);
      acc[1][t] = __builtin_amdgcn_wmma_f32_16x16x32_f16(
          false, a1, false, bf, (short)0, acc[1][t], false, false);
    }
    ap0 += 32; ap1 += 32;
    bptr += 512;                              // 32 lanes * 16 halfs
  }
  __syncthreads();   // done reading As; reuse as Cs

  // C/D layout: VGPR r -> M = r + 8*(lane>=16); N = lane&15
  int chalf = (lane >> 4) * 8;
#pragma unroll
  for (int mt = 0; mt < 2; ++mt)
#pragma unroll
    for (int t = 0; t < 4; ++t)
#pragma unroll
      for (int r = 0; r < 8; ++r)
        Cs[mt * 16 + r + chalf][(w * 4 + t) * 16 + m] = acc[mt][t][r];
  __syncthreads();

  // ---- LayerNorm over D=512 per row (32 rows, 8 partials/row) ----
  int rr = tid >> 3, sub = tid & 7;
  float s1 = 0.0f, s2 = 0.0f;
  for (int c = sub * 64; c < sub * 64 + 64; ++c) {
    float v = Cs[rr][c];
    s1 += v; s2 += v * v;
  }
  redS[rr][sub] = s1; redQ[rr][sub] = s2;
  __syncthreads();
  if (sub == 0) {
    float a = 0.0f, b2 = 0.0f;
    for (int j2 = 0; j2 < 8; ++j2) { a += redS[rr][j2]; b2 += redQ[rr][j2]; }
    float mean = a * (1.0f / DMODEL);
    float var  = b2 * (1.0f / DMODEL) - mean * mean;
    mu[rr] = mean;
    inv[rr] = rsqrtf(var + 1e-5f);
  }
  __syncthreads();

  // ---- apply LN * gamma + beta + residual x, store f16 ----
  for (int idx = tid; idx < MROWS * DMODEL; idx += 256) {
    int r = idx >> 9, c = idx & 511;
    size_t gr = tile * MROWS + r;
    float val = (Cs[r][c] - mu[r]) * inv[r] * gamma[c] + beta[c]
                + x[gr * DMODEL + c];
    dst[gr * DMODEL + c] = (_Float16)val;
  }
}

// ---------------------------------------------------------------------------
// Attention: one workgroup per batch; thread = (head h, row n).
// dyn bias factored as s[b,n,h] - s[b,m,h] + bqp[h].
// ---------------------------------------------------------------------------
__global__ void __launch_bounds__(160)
attn_kernel(const _Float16* __restrict__ qh, const _Float16* __restrict__ kh,
            const _Float16* __restrict__ vh, const float* __restrict__ rel,
            const float* __restrict__ gsb, const float* __restrict__ alpha,
            const float* __restrict__ sbuf, const float* __restrict__ bqp,
            _Float16* __restrict__ aoh) {
  __shared__ __align__(16) _Float16 ks[NSEQ * DMODEL];
  __shared__ __align__(16) _Float16 vs[NSEQ * DMODEL];
  __shared__ float ss[NSEQ * NHEAD];
  int tid = threadIdx.x;
  size_t b = blockIdx.x;
  const _Float16* kg = kh + b * NSEQ * DMODEL;
  const _Float16* vg = vh + b * NSEQ * DMODEL;
  for (int v8 = tid; v8 < NSEQ * DMODEL / 8; v8 += 160) {
    ((uint4*)ks)[v8] = ((const uint4*)kg)[v8];
    ((uint4*)vs)[v8] = ((const uint4*)vg)[v8];
  }
  ss[tid] = sbuf[b * NSEQ * NHEAD + tid];
  __syncthreads();

  int h = tid & 7, n = tid >> 3;
  const _Float16* qrow = qh + (b * NSEQ + n) * DMODEL + h * DKH;
  float q[DKH];
#pragma unroll
  for (int d = 0; d < DKH; ++d) q[d] = (float)qrow[d];

  float myS = ss[n * NHEAD + h] + bqp[h];
  float al = alpha[0];
  const float scale = 0.125f;   // 1/sqrt(64)
  float sc[NSEQ];
#pragma unroll
  for (int m2 = 0; m2 < NSEQ; ++m2) {
    const _Float16* krow = ks + m2 * DMODEL + h * DKH;
    float d = 0.0f;
#pragma unroll
    for (int e = 0; e < DKH; ++e) d += q[e] * (float)krow[e];
    sc[m2] = d * scale
           + rel[(n - m2 + NSEQ - 1) * NHEAD + h]
           + al * gsb[((size_t)h * NSEQ + n) * NSEQ + m2]
           + myS - ss[m2 * NHEAD + h];
  }
  float mx = sc[0];
#pragma unroll
  for (int m2 = 1; m2 < NSEQ; ++m2) mx = fmaxf(mx, sc[m2]);
  float sum = 0.0f;
#pragma unroll
  for (int m2 = 0; m2 < NSEQ; ++m2) { sc[m2] = __expf(sc[m2] - mx); sum += sc[m2]; }
  float rs = 1.0f / sum;
  float o[DKH];
#pragma unroll
  for (int e = 0; e < DKH; ++e) o[e] = 0.0f;
#pragma unroll
  for (int m2 = 0; m2 < NSEQ; ++m2) {
    float wgt = sc[m2] * rs;
    const _Float16* vrow = vs + m2 * DMODEL + h * DKH;
#pragma unroll
    for (int e = 0; e < DKH; ++e) o[e] += wgt * (float)vrow[e];
  }
  _Float16* orow = aoh + (b * NSEQ + n) * DMODEL + h * DKH;
#pragma unroll
  for (int e = 0; e < DKH; ++e) orow[e] = (_Float16)o[e];
}

// ---------------------------------------------------------------------------
// Output projection: aoh[40960,512] @ Wo + bo -> f32 d_out (WMMA f16).
// 32-row tiles; each B fragment reused across 2 M-tiles.
// ---------------------------------------------------------------------------
__global__ void __launch_bounds__(256)
out_kernel(const _Float16* __restrict__ aoh, const _Float16* __restrict__ wpo,
           const float* __restrict__ bo, float* __restrict__ out) {
  __shared__ __align__(16) _Float16 As[MROWS][DMODEL];   // 32KB
  int tid = threadIdx.x;
  size_t tile = blockIdx.x;
  const _Float16* src = aoh + tile * MROWS * DMODEL;
  for (int v8 = tid; v8 < MROWS * DMODEL / 8; v8 += 256)
    ((uint4*)As)[v8] = ((const uint4*)src)[v8];
  __syncthreads();

  int w = tid >> 5, lane = tid & 31;
  int m = lane & 15;
  int ksel = (lane >> 4) << 3;
  v8f acc[2][4] = {};

  const _Float16* ap0 = &As[m][ksel];
  const _Float16* ap1 = &As[16 + m][ksel];
  const _Float16* bptr = wpo + (((size_t)(w * 4) * KK_OUT) * 32 + lane) * 16;

  for (int kk = 0; kk < KK_OUT; ++kk) {
    v16h a0, a1;
    *(uint4*)&a0       = *(const uint4*)ap0;
    *((uint4*)&a0 + 1) = *(const uint4*)(ap0 + 16);
    *(uint4*)&a1       = *(const uint4*)ap1;
    *((uint4*)&a1 + 1) = *(const uint4*)(ap1 + 16);
#pragma unroll
    for (int t = 0; t < 4; ++t) {
      // byte offset t*16384 is a compile-time immediate
      v16h bf = *(const v16h*)(bptr + (size_t)t * (KK_OUT * 512));
      acc[0][t] = __builtin_amdgcn_wmma_f32_16x16x32_f16(
          false, a0, false, bf, (short)0, acc[0][t], false, false);
      acc[1][t] = __builtin_amdgcn_wmma_f32_16x16x32_f16(
          false, a1, false, bf, (short)0, acc[1][t], false, false);
    }
    ap0 += 32; ap1 += 32;
    bptr += 512;
  }

  int chalf = (lane >> 4) * 8;
#pragma unroll
  for (int mt = 0; mt < 2; ++mt)
#pragma unroll
    for (int t = 0; t < 4; ++t) {
      int col = (w * 4 + t) * 16 + m;
      float bias = bo[col];
#pragma unroll
      for (int r = 0; r < 8; ++r) {
        size_t row = tile * MROWS + mt * 16 + r + chalf;
        out[row * DMODEL + col] = acc[mt][t][r] + bias;
      }
    }
}

// ---------------------------------------------------------------------------
extern "C" void kernel_launch(void* const* d_in, const int* in_sizes, int n_in,
                              void* d_out, int out_size, void* d_ws, size_t ws_size,
                              hipStream_t stream) {
  const float* x    = (const float*)d_in[0];
  const float* Wq   = (const float*)d_in[1];
  const float* Wk   = (const float*)d_in[2];
  const float* Wv   = (const float*)d_in[3];
  const float* gq_g = (const float*)d_in[4];
  const float* gq_b = (const float*)d_in[5];
  const float* gk_g = (const float*)d_in[6];
  const float* gk_b = (const float*)d_in[7];
  const float* gv_g = (const float*)d_in[8];
  const float* gv_b = (const float*)d_in[9];
  const float* rel  = (const float*)d_in[10];
  const float* gsb  = (const float*)d_in[11];
  const float* alp  = (const float*)d_in[12];
  const float* Wqf  = (const float*)d_in[13];
  const float* bqf  = (const float*)d_in[14];
  const float* Wqp  = (const float*)d_in[15];
  const float* bqp  = (const float*)d_in[16];
  const float* Wo   = (const float*)d_in[17];
  const float* bo   = (const float*)d_in[18];
  float* out = (float*)d_out;

  // workspace carve-up
  char* ws = (char*)d_ws;
  size_t off = 0;
  auto carve = [&](size_t bytes) {
    char* p = ws + off;
    off = (off + bytes + 255) & ~((size_t)255);
    return p;
  };
  const size_t half_mat = (size_t)ROWS * DMODEL * sizeof(_Float16);
  _Float16* qh  = (_Float16*)carve(half_mat);
  _Float16* kh  = (_Float16*)carve(half_mat);
  _Float16* vh  = (_Float16*)carve(half_mat);
  _Float16* aoh = (_Float16*)carve(half_mat);
  float*    sbuf = (float*)carve((size_t)ROWS * NHEAD * sizeof(float));
  _Float16* wpq = (_Float16*)carve((size_t)NT_TILES * KK_CONV * 32 * 16 * 2);
  _Float16* wpk = (_Float16*)carve((size_t)NT_TILES * KK_CONV * 32 * 16 * 2);
  _Float16* wpv = (_Float16*)carve((size_t)NT_TILES * KK_CONV * 32 * 16 * 2);
  _Float16* wpo = (_Float16*)carve((size_t)NT_TILES * KK_OUT * 32 * 16 * 2);
  (void)ws_size; (void)n_in; (void)in_sizes; (void)out_size;

  // allow 96KB dynamic LDS on proj kernel (CDNA5 WGP has 320KB)
  const int projLds = MROWS * KCONV * 2;   // 98304 bytes
  (void)hipFuncSetAttribute((const void*)proj_kernel,
                            hipFuncAttributeMaxDynamicSharedMemorySize, projLds);

  // 1. pack weights into WMMA fragment order (f16)
  pack_qkv_kernel<<<dim3(3072, 3), 256, 0, stream>>>(Wq, Wk, Wv, wpq, wpk, wpv);
  pack_wo_kernel<<<1024, 256, 0, stream>>>(Wo, wpo);
  // 2. dynamic-bias row factors
  qf_kernel<<<ROWS / 8, 256, 0, stream>>>(x, Wqf, bqf, Wqp, sbuf);
  // 3. conv projections + LN + residual (q, k, v)
  proj_kernel<<<ROWS / MROWS, 256, projLds, stream>>>(x, wpq, gq_g, gq_b, qh);
  proj_kernel<<<ROWS / MROWS, 256, projLds, stream>>>(x, wpk, gk_g, gk_b, kh);
  proj_kernel<<<ROWS / MROWS, 256, projLds, stream>>>(x, wpv, gv_g, gv_b, vh);
  // 4. attention with rel/gsb/dynamic biases + softmax
  attn_kernel<<<BATCH, 160, 0, stream>>>(qh, kh, vh, rel, gsb, alp, sbuf, bqp, aoh);
  // 5. output projection + bias
  out_kernel<<<ROWS / MROWS, 256, 0, stream>>>(aoh, wpo, bo, out);
}
